// PersonAttention_80616536146337
// MI455X (gfx1250) — matrix operand order repaired and checked by reading, not compiled, not run on hardware
//
#include <hip/hip_runtime.h>
#include <hip/hip_bf16.h>

typedef __attribute__((ext_vector_type(16))) _Float16 v16h;
typedef __attribute__((ext_vector_type(8)))  _Float16 v8h;
typedef __attribute__((ext_vector_type(8)))  float    v8f;
typedef __attribute__((ext_vector_type(4)))  float    v4f;
typedef __attribute__((ext_vector_type(4)))  int      v4i;

#define T_DIM 128
#define B_DIM 64
#define P_DIM 16
#define F_DIM 256
#define H_DIM 8
#define D_DIM 32
#define N_ITEMS (T_DIM * B_DIM)          // 8192
#define QS 0.17677669529663689f          // 1/sqrt(32)

#if defined(__has_builtin)
#if __has_builtin(__builtin_amdgcn_global_load_async_to_lds_b128)
#define HAVE_ASYNC_LDS 1
typedef __attribute__((address_space(1))) v4i gv4i_t;   // global int4
typedef __attribute__((address_space(3))) v4i lv4i_t;   // LDS int4
#endif
#endif

// ---------------- LDS layout (bytes), total 73728 ----------------
// xf   f32 [16][256]          @ 0      (16384)  : residual copy of x
// xh   f16 [16][256]          @ 16384  ( 8192)  : x in f16 for WMMA A
// vt   f16 [8][32][32]        @ 24576  (16384)  : V transposed per head (zero padded)
// at   f16 [8][16][32]        @ 40960  ( 8192)  : attn probs, K padded to 32 with zeros
// qk   f16 [16][512]          @ 49152  (16384)  : Q|K   (aliases of)
// of   f32 [16][256]          @ 49152           : out-proj result (after qk dead)
// sc   f32 [8][16][16]        @ 65536  ( 8192)  : raw scores (aliases ch)
// ch   f16 [16][256]          @ 65536           : ctx in f16 (after sc dead)
#define SMEM_BYTES 73728

__device__ __forceinline__ v8f wmma16(v16h a, v16h b, v8f c) {
  return __builtin_amdgcn_wmma_f32_16x16x32_f16(
      /*neg_a=*/false, a, /*neg_b=*/false, b,
      /*c_mod=*/(short)0, c, /*reuse_a=*/false, /*reuse_b=*/false);
}

// Build A fragment in contiguous K order (matches B):
// lanes 0-15  : row M=lane,    K = {k0..k0+15}
// lanes 16-31 : row M=lane-16, K = {k0+16..k0+31}
// caller passes p = &buf[row*stride + k0 + hi*16]; hi = lane>>4
__device__ __forceinline__ v16h make_a(const _Float16* p) {
  v8h lo = *(const v8h*)p;
  v8h hi = *(const v8h*)(p + 8);
  return __builtin_shufflevector(lo, hi, 0,1,2,3,4,5,6,7,8,9,10,11,12,13,14,15);
}

// ---------------- weight f32 -> f16 conversion ----------------
__global__ void wconv_kernel(const float* __restrict__ win,
                             const float* __restrict__ wout,
                             _Float16* __restrict__ winh,
                             _Float16* __restrict__ wouth) {
  int i = blockIdx.x * 256 + threadIdx.x;
  if (i < 3 * F_DIM * F_DIM) winh[i]  = (_Float16)win[i];
  if (i <     F_DIM * F_DIM) wouth[i] = (_Float16)wout[i];
}

// ---------------- fused attention + residual + LN ----------------
__global__ __launch_bounds__(256)
void person_attn_kernel(const float* __restrict__ src,
                        const unsigned char* __restrict__ pmask,
                        const _Float16* __restrict__ winh,
                        const float* __restrict__ inb,
                        const _Float16* __restrict__ wouth,
                        const float* __restrict__ outb,
                        const float* __restrict__ gamma,
                        const float* __restrict__ beta,
                        float* __restrict__ out) {
  extern __shared__ char smem[];
  float*    xf = (float*)smem;
  _Float16* xh = (_Float16*)(smem + 16384);
  _Float16* vt = (_Float16*)(smem + 24576);
  _Float16* at = (_Float16*)(smem + 40960);
  _Float16* qk = (_Float16*)(smem + 49152);
  float*    of = (float*)(smem + 49152);
  float*    sc = (float*)(smem + 65536);
  _Float16* ch = (_Float16*)(smem + 65536);

  const int tid  = threadIdx.x;
  const int lane = tid & 31;
  const int wave = tid >> 5;
  const int row  = lane & 15;   // M row / N column owned by this lane
  const int hi   = lane >> 4;   // which K half-group
  const int n    = blockIdx.x;

  __builtin_prefetch(winh + (size_t)(wave * 96) * F_DIM);
  __builtin_prefetch(wouth + (size_t)(wave * 32) * F_DIM);

#ifdef HAVE_ASYNC_LDS
  // ---- Stage 1 (async path): DMA x straight into LDS (ASYNCcnt),
  //      overlap with vt/at zero-fill, then convert f32->f16 from LDS ----
  {
    const char* gbase = (const char*)(src + (size_t)n * (P_DIM * F_DIM)) + tid * 16;
    char*       lbase = (char*)xf + tid * 16;
    #pragma unroll
    for (int i = 0; i < 4; ++i) {
      __builtin_amdgcn_global_load_async_to_lds_b128(
          (gv4i_t*)(gbase + i * 4096),
          (lv4i_t*)(lbase + i * 4096),
          0, 0);
    }
    v8h z = {};
    #pragma unroll
    for (int i = 0; i < 4; ++i) ((v8h*)vt)[tid + 256 * i] = z;   // 8192 halves
    #pragma unroll
    for (int i = 0; i < 2; ++i) ((v8h*)at)[tid + 256 * i] = z;   // 4096 halves
#if __has_builtin(__builtin_amdgcn_s_wait_asynccnt)
    __builtin_amdgcn_s_wait_asynccnt(0);
#else
    asm volatile("s_wait_asynccnt 0x0" ::: "memory");
#endif
  }
  __syncthreads();
  {
    #pragma unroll
    for (int i = 0; i < 4; ++i) {
      const int base = (tid + 256 * i) * 4;
      v4f v = *(const v4f*)(xf + base);
      #pragma unroll
      for (int k = 0; k < 4; ++k) xh[base + k] = (_Float16)v[k];
    }
  }
  __syncthreads();
#else
  // ---- Stage 1 (fallback): load x via VGPRs, write f32 + f16 copies ----
  {
    const v4f* g = (const v4f*)(src + (size_t)n * (P_DIM * F_DIM));
    #pragma unroll
    for (int i = 0; i < 4; ++i) {
      v4f v = g[tid + 256 * i];
      int base = (tid + 256 * i) * 4;
      #pragma unroll
      for (int k = 0; k < 4; ++k) {
        xf[base + k] = v[k];
        xh[base + k] = (_Float16)v[k];
      }
    }
    v8h z = {};
    #pragma unroll
    for (int i = 0; i < 4; ++i) ((v8h*)vt)[tid + 256 * i] = z;   // 8192 halves
    #pragma unroll
    for (int i = 0; i < 2; ++i) ((v8h*)at)[tid + 256 * i] = z;   // 4096 halves
  }
  __syncthreads();
#endif

  // ---- Stage 2: QKV = x @ Win^T + b ; 48 column tiles, 6 per wave ----
  #pragma unroll
  for (int q6 = 0; q6 < 6; ++q6) {
    const int jt   = wave * 6 + q6;
    const int jcol = jt * 16 + row;
    v8f acc = {};
    #pragma unroll
    for (int kk = 0; kk < 8; ++kk) {
      const int k0 = kk * 32;
      v16h a = make_a(xh + row * 256 + k0 + hi * 16);
      v16h b = *(const v16h*)(winh + (size_t)jcol * 256 + k0 + hi * 16);
      acc = wmma16(a, b, acc);
    }
    const float bias = inb[jcol];
    const float qs   = (jt < 16) ? QS : 1.0f;   // scale only Q columns
    if (jt < 32) {                               // Q or K -> qk[p][jcol]
      #pragma unroll
      for (int r = 0; r < 8; ++r)
        qk[(r + 8 * hi) * 512 + jcol] = (_Float16)((acc[r] + bias) * qs);
    } else {                                     // V -> transposed vt[h][d][p]
      const int h  = (jcol - 512) >> 5;
      const int dd = (jcol - 512) & 31;
      #pragma unroll
      for (int r = 0; r < 8; ++r)
        vt[h * 1024 + dd * 32 + (r + 8 * hi)] = (_Float16)(acc[r] + bias);
    }
  }
  __syncthreads();

  // ---- Stage 3: scores[h] = q_h @ k_h^T  (one WMMA per head, wave==head) ----
  {
    const int h = wave;
    v16h a = make_a(qk + row * 512 + h * 32 + hi * 16);               // q rows
    v16h b = *(const v16h*)(qk + row * 512 + 256 + h * 32 + hi * 16); // k rows
    v8f acc = {};
    acc = wmma16(a, b, acc);
    #pragma unroll
    for (int r = 0; r < 8; ++r)
      sc[h * 256 + (r + 8 * hi) * 16 + row] = acc[r];
  }
  __syncthreads();

  // ---- Stage 4: masked softmax over 17 keys (16 real + zero token) ----
  if (lane < 16) {
    const int h = wave;
    const unsigned char* pm = pmask + (size_t)n * P_DIM;
    float s[16];
    float m = 0.0f;                        // zero-token score is 0, never masked
    #pragma unroll
    for (int k = 0; k < 16; ++k) {
      float v = sc[h * 256 + lane * 16 + k];
      if (pm[k]) v = -1e9f;
      s[k] = v;
      m = fmaxf(m, v);
    }
    float sum = __expf(0.0f - m);          // zero token contribution
    #pragma unroll
    for (int k = 0; k < 16; ++k) { float e = __expf(s[k] - m); s[k] = e; sum += e; }
    const float ri = 1.0f / sum;
    #pragma unroll
    for (int k = 0; k < 16; ++k)
      at[h * 512 + lane * 32 + k] = (_Float16)(s[k] * ri);
  }
  __syncthreads();

  // ---- Stage 5: ctx_h = attn_h @ v_h  (2 WMMAs per head) ----
  {
    const int h = wave;
    v16h a = make_a(at + h * 512 + row * 32 + hi * 16);
    #pragma unroll
    for (int dt = 0; dt < 2; ++dt) {
      v16h b = *(const v16h*)(vt + h * 1024 + (dt * 16 + row) * 32 + hi * 16);
      v8f acc = {};
      acc = wmma16(a, b, acc);
      #pragma unroll
      for (int r = 0; r < 8; ++r)
        ch[(r + 8 * hi) * 256 + h * 32 + dt * 16 + row] = (_Float16)acc[r];
    }
  }
  __syncthreads();

  // ---- Stage 6: out = ctx @ Wout^T + b ; 16 tiles, 2 per wave ----
  #pragma unroll
  for (int q2 = 0; q2 < 2; ++q2) {
    const int jt   = wave * 2 + q2;
    const int jcol = jt * 16 + row;
    v8f acc = {};
    #pragma unroll
    for (int kk = 0; kk < 8; ++kk) {
      const int k0 = kk * 32;
      v16h a = make_a(ch + row * 256 + k0 + hi * 16);
      v16h b = *(const v16h*)(wouth + (size_t)jcol * 256 + k0 + hi * 16);
      acc = wmma16(a, b, acc);
    }
    const float bias = outb[jcol];
    #pragma unroll
    for (int r = 0; r < 8; ++r)
      of[(r + 8 * hi) * 256 + jcol] = acc[r] + bias;
  }
  __syncthreads();

  // ---- Stage 7: residual + LayerNorm; 16 lanes cooperate per row ----
  {
    const int p   = tid >> 4;
    const int sub = tid & 15;
    float y[16];
    float s1 = 0.0f, s2 = 0.0f;
    #pragma unroll
    for (int i = 0; i < 16; ++i) {
      const int c = i * 16 + sub;
      float v = xf[p * 256 + c] + of[p * 256 + c];
      y[i] = v; s1 += v; s2 += v * v;
    }
    #pragma unroll
    for (int m = 1; m < 16; m <<= 1) {
      s1 += __shfl_xor(s1, m, 32);
      s2 += __shfl_xor(s2, m, 32);
    }
    const float mu  = s1 * (1.0f / 256.0f);
    const float var = s2 * (1.0f / 256.0f) - mu * mu;
    const float ri  = rsqrtf(var + 1e-5f);
    float* o = out + (size_t)n * (P_DIM * F_DIM) + p * 256;
    #pragma unroll
    for (int i = 0; i < 16; ++i) {
      const int c = i * 16 + sub;
      o[c] = gamma[c] * (y[i] - mu) * ri + beta[c];
    }
  }
}

extern "C" void kernel_launch(void* const* d_in, const int* in_sizes, int n_in,
                              void* d_out, int out_size, void* d_ws, size_t ws_size,
                              hipStream_t stream) {
  const float*         src   = (const float*)d_in[0];
  const unsigned char* pmask = (const unsigned char*)d_in[1];   // numpy bool
  const float*         winf  = (const float*)d_in[2];
  const float*         inb   = (const float*)d_in[3];
  const float*         woutf = (const float*)d_in[4];
  const float*         outb  = (const float*)d_in[5];
  const float*         gamma = (const float*)d_in[6];
  const float*         beta  = (const float*)d_in[7];

  _Float16* winh  = (_Float16*)d_ws;                // 3*256*256 halves
  _Float16* wouth = winh + 3 * F_DIM * F_DIM;       // 256*256 halves

  wconv_kernel<<<(3 * F_DIM * F_DIM + 255) / 256, 256, 0, stream>>>(
      winf, woutf, winh, wouth);

  person_attn_kernel<<<N_ITEMS, 256, SMEM_BYTES, stream>>>(
      src, pmask, winh, inb, wouth, outb, gamma, beta, (float*)d_out);
}
